// ssm_68968584839901
// MI455X (gfx1250) — matrix operand order repaired and checked by reading, not compiled, not run on hardware
//
#include <hip/hip_runtime.h>

// ---------------------------------------------------------------------------
// Mamba (selective scan) forward for MI455X / gfx1250.
//  - All four GEMMs on v_wmma_f32_16x16x32_bf16 (f32 accumulate)
//  - GEMM tiles staged LDS-direct via the Tensor Data Mover (tensor_load_to_lds),
//    double-buffered on TENSORcnt so DMA overlaps WMMA compute
//  - wave32 scan: one lane per (b,d) channel, B/C vectors chunked through LDS
//   B=2, L=2048, D_MODEL=1024, D_INNER=2048, D_STATE=16, D_CONV=4, DT_RANK=64
// ---------------------------------------------------------------------------

typedef unsigned short ushort_t;
typedef unsigned int   uint_t;

typedef __attribute__((ext_vector_type(8)))  float        v8f;
typedef __attribute__((ext_vector_type(16))) __bf16       v16bf;
typedef __attribute__((ext_vector_type(8)))  __bf16       v8bf;
typedef __attribute__((ext_vector_type(4)))  unsigned int v4u;
typedef __attribute__((ext_vector_type(8)))  int          v8i;
typedef __attribute__((ext_vector_type(4)))  int          v4i;

#define B_      2
#define L_      2048
#define DM_     1024
#define DI_     2048
#define N2_     4096      // 2 * D_INNER
#define DSTATE_ 16
#define DCONV_  4
#define DTR_    64
#define GDIM_   96        // DT_RANK + 2*D_STATE
#define MROWS_  4096      // B_ * L_

// ------------------------------ helpers ------------------------------------

__device__ __forceinline__ ushort_t f32_to_bf16(float f) {
  uint_t x = __float_as_uint(f);
  if ((x & 0x7F800000u) == 0x7F800000u) {
    uint_t r = x >> 16;
    if (x & 0x007FFFFFu) r |= 0x40u;     // keep NaN a NaN
    return (ushort_t)r;
  }
  uint_t r = x + 0x7FFFu + ((x >> 16) & 1u);   // round-to-nearest-even
  return (ushort_t)(r >> 16);
}

__device__ __forceinline__ float sigmoidf_(float x) {
  return 1.0f / (1.0f + __expf(-x));
}

// --------------------- Tensor Data Mover 2D tile load ----------------------
// Loads a (128 rows x 64 elems) bf16 tile, row stride = stride_elems, from
// global 'gptr' (tile origin) into LDS byte offset 'lds_off'. Rows beyond
// rows_rem are zero-filled by the TDM OOB rule (ISA 08_async_tensor §8.2).

__device__ __forceinline__ void tdm_load_tile(unsigned lds_off, const void* gptr,
                                              unsigned rows_rem, unsigned k_rem,
                                              unsigned stride_elems)
{
  unsigned long long ga = (unsigned long long)(size_t)gptr;
  v4u g0;
  g0.x = 1u;                                   // count=1, user descriptor
  g0.y = lds_off;                              // lds_addr (bytes)
  g0.z = (unsigned)(ga & 0xFFFFFFFFull);       // global_addr[31:0]
  g0.w = (unsigned)(ga >> 32) | (2u << 30);    // global_addr[56:32] | type=2
  v8i g1;
  g1[0] = (int)(1u << 16);                     // wg_mask=0, data_size=1 (2B)
  g1[1] = (int)((k_rem & 0xFFFFu) << 16);      // tensor_dim0[15:0]
  g1[2] = (int)((k_rem >> 16) | ((rows_rem & 0xFFFFu) << 16)); // td0 hi | td1 lo
  g1[3] = (int)((rows_rem >> 16) | (64u << 16));               // td1 hi | tile_dim0=64
  g1[4] = (int)(128u);                         // tile_dim1=128, tile_dim2=0
  g1[5] = (int)(stride_elems);                 // tensor_dim0_stride[31:0]
  g1[6] = 0;                                   // stride hi, dim1_stride lo
  g1[7] = 0;
  v4i z4 = {0, 0, 0, 0};                       // groups 2/3 unused (2D tensor)
#if defined(__clang_major__) && (__clang_major__ >= 23)
  v8i z8 = {0, 0, 0, 0, 0, 0, 0, 0};
  __builtin_amdgcn_tensor_load_to_lds(g0, g1, z4, z4, z8, 0);
#else
  __builtin_amdgcn_tensor_load_to_lds(g0, g1, z4, z4, 0);
#endif
}

// ------------------------------ cast kernels -------------------------------

__global__ void cast_f32_to_bf16_k(const float* __restrict__ in,
                                   ushort_t* __restrict__ out, int n) {
  int i = blockIdx.x * blockDim.x + threadIdx.x;
  if (i < n) out[i] = f32_to_bf16(in[i]);
}

// dt_low slice (cols [0,64) of x_dbl (M x 96)) -> bf16 (M x 64)
__global__ void cast_dtlow_k(const float* __restrict__ xdbl,
                             ushort_t* __restrict__ out, int n) {
  int i = blockIdx.x * blockDim.x + threadIdx.x;
  if (i < n) {
    int row = i >> 6;
    int c   = i & 63;
    out[i] = f32_to_bf16(xdbl[(size_t)row * GDIM_ + c]);
  }
}

// ------------------------------ WMMA GEMM ----------------------------------
// C(MxN, f32, row-major) = A(MxK, bf16, row-major) * W(NxK, bf16, row-major)^T
// TDM double-buffered tile staging; mode 1 fuses softplus(v + bias[col]).

#define BM 128
#define BN 128
#define BK 64

union V16U { v16bf v; v8bf h[2]; };

__global__ __launch_bounds__(256)
void gemm_bf16_tn_k(const ushort_t* __restrict__ A, const ushort_t* __restrict__ W,
                    float* __restrict__ C, const float* __restrict__ bias,
                    int M, int N, int K, int mode)
{
  __shared__ __align__(16) ushort_t sA[2][BM * BK];   // 2 x 16 KB
  __shared__ __align__(16) ushort_t sB[2][BN * BK];   // 2 x 16 KB

  const int tid  = threadIdx.x;
  const int lane = tid & 31;
  const int wave = tid >> 5;
  const int p    = lane & 15;     // fragment row (A) / col (B)
  const int hi   = lane >> 4;     // half-wave select

  const int m0 = blockIdx.y * BM;
  const int n0 = blockIdx.x * BN;
  const int wm = (wave & 3) << 5;   // wave M-offset: 0,32,64,96
  const int wn = (wave >> 2) << 6;  // wave N-offset: 0,64

  const unsigned ldsA[2] = { (unsigned)(size_t)&sA[0][0], (unsigned)(size_t)&sA[1][0] };
  const unsigned ldsB[2] = { (unsigned)(size_t)&sB[0][0], (unsigned)(size_t)&sB[1][0] };

  const unsigned remA = (unsigned)(M - m0);   // rows available from tile origin
  const unsigned remB = (unsigned)(N - n0);   // < 128 for x_proj -> TDM zero-fill
  const int nt = K / BK;

  v8f acc[2][4] = {};

  // Prologue: wave 0 issues the DMA pair for tile 0 (TDM ignores EXEC; only
  // wave 0's instruction stream reaches it).
  if (tid < 32) {
    tdm_load_tile(ldsA[0], A + (size_t)m0 * K, remA, (unsigned)K, (unsigned)K);
    tdm_load_tile(ldsB[0], W + (size_t)n0 * K, remB, (unsigned)K, (unsigned)K);
  }

  for (int t = 0; t < nt; ++t) {
    __syncthreads();   // everyone done computing from the buffer loaded next
    if (tid < 32) {
      if (t + 1 < nt) {
        const size_t koff = (size_t)(t + 1) * BK;
        const int nb = (t + 1) & 1;
        tdm_load_tile(ldsA[nb], A + (size_t)m0 * K + koff, remA,
                      (unsigned)(K - (int)koff), (unsigned)K);
        tdm_load_tile(ldsB[nb], W + (size_t)n0 * K + koff, remB,
                      (unsigned)(K - (int)koff), (unsigned)K);
        // pairs complete in order: <=2 outstanding => tile t's pair landed
        __builtin_amdgcn_s_wait_tensorcnt(2);
      } else {
        __builtin_amdgcn_s_wait_tensorcnt(0);
      }
    }
    __syncthreads();   // tile t visible to all waves

    const ushort_t* tA = &sA[t & 1][0];
    const ushort_t* tB = &sB[t & 1][0];

#pragma unroll
    for (int kk = 0; kk < 2; ++kk) {          // two WMMA K-steps per DMA tile
      const int ko = kk << 5;
      // A fragments: lanes<16 take K[0..8)+K[16..24), lanes>=16 shifted +8
      v16bf af[2];
#pragma unroll
      for (int fm = 0; fm < 2; ++fm) {
        V16U u;
        const int base = (wm + (fm << 4) + p) * BK + ko + (hi << 3);
        u.h[0] = *(const v8bf*)&tA[base];
        u.h[1] = *(const v8bf*)&tA[base + 16];
        af[fm] = u.v;
      }
      // B fragments: lane p = column; lanes<16 take K[0..8)+K[16..24), lanes>=16 shifted +8
      v16bf bfr[4];
#pragma unroll
      for (int fn = 0; fn < 4; ++fn) {
        V16U u;
        const int base = (wn + (fn << 4) + p) * BK + ko + (hi << 3);
        u.h[0] = *(const v8bf*)&tB[base];
        u.h[1] = *(const v8bf*)&tB[base + 16];
        bfr[fn] = u.v;
      }
#pragma unroll
      for (int fm = 0; fm < 2; ++fm)
#pragma unroll
        for (int fn = 0; fn < 4; ++fn)
          acc[fm][fn] = __builtin_amdgcn_wmma_f32_16x16x32_bf16(
              false, af[fm], false, bfr[fn], (short)0, acc[fm][fn], false, false);
    }
  }

  // Epilogue: C/D layout -> lane holds col (lane&15), rows hi*8 + r.
#pragma unroll
  for (int fm = 0; fm < 2; ++fm) {
    const int rbase = m0 + wm + (fm << 4) + (hi << 3);
#pragma unroll
    for (int fn = 0; fn < 4; ++fn) {
      const int col = n0 + wn + (fn << 4) + p;
      if (col >= N) continue;
#pragma unroll
      for (int r = 0; r < 8; ++r) {
        const int row = rbase + r;
        if (row >= M) continue;
        float v = acc[fm][fn][r];
        if (mode == 1) {
          v += bias[col];
          v = (v > 20.f) ? v : log1pf(__expf(v));   // softplus
        }
        C[(size_t)row * N + col] = v;
      }
    }
  }
}

// -------------------- depthwise causal conv (k=4) + SiLU -------------------

__global__ __launch_bounds__(256)
void conv_silu_k(const float* __restrict__ xz, const float* __restrict__ conv_w,
                 const float* __restrict__ conv_b,
                 float* __restrict__ xc, ushort_t* __restrict__ xc_bf)
{
  int idx = blockIdx.x * blockDim.x + threadIdx.x;     // over MROWS_*DI_
  if (idx >= MROWS_ * DI_) return;
  int e   = idx & (DI_ - 1);
  int row = idx >> 11;           // / DI_
  int l   = row & (L_ - 1);
  int b   = row >> 11;           // / L_
  float s = conv_b[e];
#pragma unroll
  for (int k = 0; k < DCONV_; ++k) {
    int lp = l - (DCONV_ - 1) + k;
    if (lp >= 0)
      s += xz[((size_t)(b * L_ + lp)) * N2_ + e] * conv_w[e * DCONV_ + k];
  }
  float v = s * sigmoidf_(s);
  xc[idx]    = v;
  xc_bf[idx] = f32_to_bf16(v);
}

// ------------------------------ selective scan -----------------------------
// One lane per (b, d) channel; 16-state recurrence over L. B/C vectors for
// 64 timesteps staged through LDS per chunk. Fuses +u*D, *silu(z), bf16 out.

__global__ __launch_bounds__(256)
void scan_k(const float* __restrict__ delta, const float* __restrict__ xc,
            const float* __restrict__ xdbl,  const float* __restrict__ xz,
            const float* __restrict__ A_log, const float* __restrict__ D_param,
            ushort_t* __restrict__ y_bf)
{
  __shared__ float sBC[64 * 2 * DSTATE_];   // 8 KB: [t][B(16) C(16)]

  const int tid = threadIdx.x;
  const int blk = blockIdx.x;               // 16 blocks
  const int b   = blk >> 3;
  const int d   = ((blk & 7) << 8) + tid;

  float Arow[DSTATE_], st[DSTATE_];
#pragma unroll
  for (int n = 0; n < DSTATE_; ++n) {
    Arow[n] = -__expf(A_log[d * DSTATE_ + n]);
    st[n]   = 0.f;
  }
  const float Dd = D_param[d];

  for (int t0 = 0; t0 < L_; t0 += 64) {
    __syncthreads();
#pragma unroll
    for (int i = 0; i < 8; ++i) {           // 2048 floats / 256 threads
      int c  = tid + (i << 8);
      int tt = c >> 5, j = c & 31;
      sBC[c] = xdbl[((size_t)(b * L_ + t0 + tt)) * GDIM_ + DTR_ + j];
    }
    __syncthreads();

    for (int ti = 0; ti < 64; ++ti) {
      const size_t row = (size_t)(b * L_ + t0 + ti);
      const float dt = delta[row * DI_ + d];
      const float u  = xc[row * DI_ + d];
      const float du = dt * u;
      const float* bc = &sBC[ti * 32];
      float y = 0.f;
#pragma unroll
      for (int n = 0; n < DSTATE_; ++n) {
        float dA = __expf(dt * Arow[n]);
        float s  = dA * st[n] + du * bc[n];
        st[n] = s;
        y += s * bc[DSTATE_ + n];
      }
      const float z  = xz[row * N2_ + DI_ + d];
      const float yv = (y + u * Dd) * (z * sigmoidf_(z));
      y_bf[row * DI_ + d] = f32_to_bf16(yv);
    }
  }
}

// ------------------------------ host launcher ------------------------------

extern "C" void kernel_launch(void* const* d_in, const int* in_sizes, int n_in,
                              void* d_out, int out_size, void* d_ws, size_t ws_size,
                              hipStream_t stream) {
  (void)in_sizes; (void)n_in; (void)out_size; (void)ws_size;

  const float* hs     = (const float*)d_in[0];   // (B,L,DM)
  const float* w_in   = (const float*)d_in[1];   // (N2,DM)
  const float* conv_w = (const float*)d_in[2];   // (DI,1,4)
  const float* conv_b = (const float*)d_in[3];   // (DI)
  const float* w_xp   = (const float*)d_in[4];   // (96,DI)
  const float* w_dt   = (const float*)d_in[5];   // (DI,64)
  const float* b_dt   = (const float*)d_in[6];   // (DI)
  const float* A_log  = (const float*)d_in[7];   // (DI,16)
  const float* D_par  = (const float*)d_in[8];   // (DI)
  const float* w_out  = (const float*)d_in[9];   // (DM,DI)
  float* out = (float*)d_out;                    // (B,L,DM)

  // ---- workspace layout (bytes) ----
  char* ws = (char*)d_ws;
  size_t o = 0;
  float*    XZ    = (float*)(ws + o); o += (size_t)MROWS_ * N2_  * 4;  // 64 MB
  float*    XC    = (float*)(ws + o); o += (size_t)MROWS_ * DI_  * 4;  // 32 MB
  float*    DELTA = (float*)(ws + o); o += (size_t)MROWS_ * DI_  * 4;  // 32 MB
  float*    XDBL  = (float*)(ws + o); o += (size_t)MROWS_ * GDIM_* 4;
  ushort_t* HSB   = (ushort_t*)(ws + o); o += (size_t)MROWS_ * DM_ * 2;
  ushort_t* WINB  = (ushort_t*)(ws + o); o += (size_t)N2_    * DM_ * 2;
  ushort_t* WXPB  = (ushort_t*)(ws + o); o += (size_t)GDIM_  * DI_ * 2;
  ushort_t* WDTB  = (ushort_t*)(ws + o); o += (size_t)DI_    * DTR_* 2;
  ushort_t* WOUTB = (ushort_t*)(ws + o); o += (size_t)DM_    * DI_ * 2;
  ushort_t* XCB   = (ushort_t*)(ws + o); o += (size_t)MROWS_ * DI_ * 2;
  ushort_t* DTB   = (ushort_t*)(ws + o); o += (size_t)MROWS_ * DTR_* 2;
  ushort_t* YB    = (ushort_t*)(ws + o); o += (size_t)MROWS_ * DI_ * 2;

  const int T = 256;
  auto cdiv = [](int a, int b) { return (a + b - 1) / b; };

  // 1) bf16 casts of activations + weights
  {
    int n;
    n = MROWS_ * DM_;  cast_f32_to_bf16_k<<<cdiv(n,T), T, 0, stream>>>(hs,    HSB,   n);
    n = N2_ * DM_;     cast_f32_to_bf16_k<<<cdiv(n,T), T, 0, stream>>>(w_in,  WINB,  n);
    n = GDIM_ * DI_;   cast_f32_to_bf16_k<<<cdiv(n,T), T, 0, stream>>>(w_xp,  WXPB,  n);
    n = DI_ * DTR_;    cast_f32_to_bf16_k<<<cdiv(n,T), T, 0, stream>>>(w_dt,  WDTB,  n);
    n = DM_ * DI_;     cast_f32_to_bf16_k<<<cdiv(n,T), T, 0, stream>>>(w_out, WOUTB, n);
  }

  // 2) in_proj: XZ(M x 4096) = HS(M x 1024) * Win(4096 x 1024)^T
  gemm_bf16_tn_k<<<dim3(cdiv(N2_,BN), cdiv(MROWS_,BM)), T, 0, stream>>>(
      HSB, WINB, XZ, nullptr, MROWS_, N2_, DM_, 0);

  // 3) depthwise causal conv + SiLU -> XC (+ bf16 copy)
  conv_silu_k<<<cdiv(MROWS_ * DI_, T), T, 0, stream>>>(XZ, conv_w, conv_b, XC, XCB);

  // 4) x_proj: XDBL(M x 96) = XC(M x 2048) * Wxp(96 x 2048)^T  (TDM zero-fills rows 96..127)
  gemm_bf16_tn_k<<<dim3(cdiv(GDIM_,BN), cdiv(MROWS_,BM)), T, 0, stream>>>(
      XCB, WXPB, XDBL, nullptr, MROWS_, GDIM_, DI_, 0);

  // 5) dt_low slice -> bf16
  cast_dtlow_k<<<cdiv(MROWS_ * DTR_, T), T, 0, stream>>>(XDBL, DTB, MROWS_ * DTR_);

  // 6) dt_proj + softplus: DELTA(M x 2048) = softplus(DTB * Wdt^T + b_dt)
  gemm_bf16_tn_k<<<dim3(cdiv(DI_,BN), cdiv(MROWS_,BM)), T, 0, stream>>>(
      DTB, WDTB, DELTA, b_dt, MROWS_, DI_, DTR_, 1);

  // 7) selective scan + skip + gate -> YB (bf16)
  scan_k<<<B_ * (DI_ / T), T, 0, stream>>>(DELTA, XC, XDBL, XZ, A_log, D_par, YB);

  // 8) out_proj: OUT(M x 1024) = YB(M x 2048) * Wout(1024 x 2048)^T
  gemm_bf16_tn_k<<<dim3(cdiv(DM_,BN), cdiv(MROWS_,BM)), T, 0, stream>>>(
      YB, WOUTB, out, nullptr, MROWS_, DM_, DI_, 0);
}
